// SimplicialAttentionLayer2_16655883174240
// MI455X (gfx1250) — matrix operand order repaired and checked by reading, not compiled, not run on hardware
//
#include <hip/hip_runtime.h>
#include <math.h>

// ---------------------------------------------------------------------------
// SimplicialAttentionLayer2 for MI455X (gfx1250, wave32, WMMA bf16 path)
//   N = 4096 simplices, D = 1024 features
//   All GEMMs: NT bf16 WMMA (v_wmma_f32_16x16x32_bf16), 32x64 register tile
//   per wave, 128x128 block tile staged through LDS by the Tensor Data Mover
//   (tensor_load_to_lds + s_wait_tensorcnt, double buffered), padded LDS
//   pitch (144B) for conflict-free ds_load_b128 fragment reads.
// ---------------------------------------------------------------------------

#define NSIMP 4096
#define DFEAT 1024

typedef __attribute__((ext_vector_type(16))) __bf16         v16bf;
typedef __attribute__((ext_vector_type(8)))  float          v8f;
typedef __attribute__((ext_vector_type(8)))  unsigned short ushort8;
typedef __attribute__((ext_vector_type(4)))  unsigned int   u32x4;
typedef __attribute__((ext_vector_type(8)))  int            i32x8;
typedef __attribute__((ext_vector_type(4)))  int            i32x4;

#ifndef __has_builtin
#define __has_builtin(x) 0
#endif
#if defined(__AMDGCN__) && __has_builtin(__builtin_amdgcn_tensor_load_to_lds)
#define USE_TDM 1
#else
#define USE_TDM 0
#endif

union FragAB {
    v16bf          bf;
    ushort8        h2[2];
    unsigned short u[16];
};

// fp32 -> bf16 round-to-nearest-even
__device__ __forceinline__ unsigned short f2bf_rn(float f) {
    union { float f; unsigned int u; } v; v.f = f;
    unsigned int r = v.u + 0x7FFFu + ((v.u >> 16) & 1u);
    return (unsigned short)(r >> 16);
}

// LDS staging geometry: 128 rows x 64 bf16, padded to 72-element (144 B) pitch
#define KC        64
#define LDS_PITCH 72
#define LDS_BUF_ELems (128 * LDS_PITCH)
#define LDS_BUF_BYTES (LDS_BUF_ELems * 2)

#if USE_TDM
// ---------------------------------------------------------------------------
// Issue one TDM 2-D tile load: 128 rows x 64 bf16 from row-major [rows, K]
// global matrix into LDS at byte offset lds_off, with pad_amount=4 DWORDs
// after every 32 DWORDs (one 128B row) -> 144B LDS pitch.
// D# layout per CDNA5 ISA 8.3-8.6. Wave-level op; call from one wave only.
// (6-arg toolchain variant: g0, g1, g2, g3, extra, cpol)
// ---------------------------------------------------------------------------
__device__ __forceinline__ void tdm_load_tile(unsigned int lds_off,
                                              const unsigned short* gptr,
                                              unsigned int K) {
    unsigned long long ga = (unsigned long long)gptr;
    u32x4 g0;
    g0[0] = 1u;                                        // count=1, load, user D#
    g0[1] = lds_off;                                   // lds_addr (bytes)
    g0[2] = (unsigned int)(ga & 0xFFFFFFFFu);          // global_addr[31:0]
    g0[3] = (unsigned int)((ga >> 32) & 0x01FFFFFFu)   // global_addr[56:32]
          | (2u << 30);                                // type = 2 ("image")
    i32x8 g1;
    g1[0] = (int)((1u << 16)      // data_size = 1 -> 2 bytes
                | (1u << 20)      // pad_enable
                | (4u << 22)      // pad_interval: 1<<4 * 8B = 128B rows
                | (3u << 25));    // pad_amount: 4 DWORDs = 16B
    g1[1] = (int)((K & 0xFFFFu) << 16);                       // tensor_dim0 lo
    g1[2] = (int)(((K >> 16) & 0xFFFFu) | (128u << 16));      // td0 hi | td1 lo
    g1[3] = (int)((0u) | ((unsigned)KC << 16));               // td1 hi | tile_dim0
    g1[4] = (int)(128u);                                      // tile_dim1 (td2=0)
    g1[5] = (int)K;                                           // td0_stride[31:0]
    g1[6] = 0;                                                // stride hi, td1_stride
    g1[7] = 0;
    i32x4 z4 = {0, 0, 0, 0};
    i32x8 z8 = {0, 0, 0, 0, 0, 0, 0, 0};
    __builtin_amdgcn_tensor_load_to_lds(g0, g1, z4, z4, z8, 0);
}
#endif

// ---------------------------------------------------------------------------
// Elementwise fp32 -> bf16 conversion (grid-stride)
// ---------------------------------------------------------------------------
__global__ void cvt_f32_to_bf16(const float* __restrict__ in,
                                unsigned short* __restrict__ out, int n) {
    int i = blockIdx.x * blockDim.x + threadIdx.x;
    int stride = gridDim.x * blockDim.x;
    for (; i < n; i += stride) out[i] = f2bf_rn(in[i]);
}

// ---------------------------------------------------------------------------
// Tiled transpose + fp32->bf16 convert: in [R,C] f32 -> out [C,R] bf16.
// ---------------------------------------------------------------------------
__global__ void transpose_cvt_bf16(const float* __restrict__ in,
                                   unsigned short* __restrict__ out,
                                   int R, int C) {
    __shared__ unsigned short tile[32][33];
    const int tilesC = C >> 5;
    const int bx = blockIdx.x % tilesC;
    const int by = blockIdx.x / tilesC;
    const int tx = threadIdx.x & 31;
    const int ty = threadIdx.x >> 5;
#pragma unroll
    for (int i = 0; i < 4; ++i) {
        int r = by * 32 + ty + 8 * i;
        tile[ty + 8 * i][tx] = f2bf_rn(in[(size_t)r * C + bx * 32 + tx]);
    }
    __syncthreads();
#pragma unroll
    for (int i = 0; i < 4; ++i) {
        int r = bx * 32 + ty + 8 * i;
        out[(size_t)r * R + by * 32 + tx] = tile[tx][ty + 8 * i];
    }
}

// ---------------------------------------------------------------------------
// Transposed nonzero mask: Mt[i,j] = (B[j,i] != 0), byte output.
// ---------------------------------------------------------------------------
__global__ void mask_transpose(const float* __restrict__ B,
                               unsigned char* __restrict__ Mt, int n) {
    __shared__ unsigned char tile[32][33];
    const int tiles = n >> 5;
    const int bx = blockIdx.x % tiles;
    const int by = blockIdx.x / tiles;
    const int tx = threadIdx.x & 31;
    const int ty = threadIdx.x >> 5;
#pragma unroll
    for (int i = 0; i < 4; ++i) {
        int r = by * 32 + ty + 8 * i;
        tile[ty + 8 * i][tx] = (B[(size_t)r * n + bx * 32 + tx] != 0.0f) ? 1 : 0;
    }
    __syncthreads();
#pragma unroll
    for (int i = 0; i < 4; ++i) {
        int r = bx * 32 + ty + 8 * i;
        Mt[(size_t)r * n + by * 32 + tx] = tile[tx][ty + 8 * i];
    }
}

// ---------------------------------------------------------------------------
// NT WMMA bf16 GEMM: C[m,n] = sum_k A[m,k] * B[n,k]
//   A: [M,K] bf16 row-major, B: [Ncols,K] bf16 row-major, K % 64 == 0.
//   Block: 256 threads = 8 waves (4 M x 2 N); block tile 128x128.
//   Wave tile: 32x64 (2x4 accumulators). K chunks of 64 staged into LDS by
//   the TDM (double buffered), fragments read with ds_load_b128.
//   EPI 0: C fp32 = alpha*acc   EPI 1: C bf16 = acc   EPI 2: fp32 tanh(+bias)
// ---------------------------------------------------------------------------
template <int EPI>
__global__ void gemm_nt_wmma_bf16(const unsigned short* __restrict__ A,
                                  const unsigned short* __restrict__ B,
                                  void* __restrict__ C,
                                  const float* __restrict__ bias,
                                  int K, int ldc, int c_col_off, float alpha) {
    const int lane = threadIdx.x & 31;
    const int kh   = lane >> 4;        // lane-half select (0/1)
    const int l16  = lane & 15;
    const int wid  = threadIdx.x >> 5; // 0..7
    const int rowblk = blockIdx.y * 128;
    const int colblk = blockIdx.x * 128;
    const int warow  = (wid >> 1) * 32; // wave row offset inside block tile
    const int wacol  = (wid & 1) * 64;  // wave col offset inside block tile

    // sm[0..1]: A double buffer, sm[2..3]: B double buffer (72 KB total).
    // Single shared object -> buffer b sits at LDS byte offset b*LDS_BUF_BYTES.
    __shared__ __align__(16) unsigned short sm[4][LDS_BUF_ELems];

    v8f acc[2][4] = {};
    const int nk = K / KC;

#if USE_TDM
    if (wid == 0) {
        tdm_load_tile(0u * LDS_BUF_BYTES, A + (size_t)rowblk * K, (unsigned)K);
        tdm_load_tile(2u * LDS_BUF_BYTES, B + (size_t)colblk * K, (unsigned)K);
    }
#endif

    for (int ki = 0; ki < nk; ++ki) {
#if USE_TDM
        const int cur = ki & 1;
        if (wid == 0) {
            if (ki + 1 < nk) {
                const unsigned nxt = (unsigned)(cur ^ 1);
                const size_t kofs = (size_t)(ki + 1) * KC;
                tdm_load_tile(nxt * LDS_BUF_BYTES,
                              A + (size_t)rowblk * K + kofs, (unsigned)K);
                tdm_load_tile((2u + nxt) * LDS_BUF_BYTES,
                              B + (size_t)colblk * K + kofs, (unsigned)K);
                __builtin_amdgcn_s_wait_tensorcnt(2); // current buffers done
            } else {
                __builtin_amdgcn_s_wait_tensorcnt(0);
            }
        }
        __syncthreads();               // publish buffer `cur` to all waves
#else
        const int cur = 0;
        __syncthreads();               // protect previous iteration's reads
        // Cooperative global->LDS copy fallback: 128x64 bf16 per operand,
        // 1024 ushort8 chunks per operand, 4 per thread.
#pragma unroll
        for (int c = 0; c < 4; ++c) {
            int idx = threadIdx.x + c * 256;
            int r = idx >> 3, kcol = (idx & 7) * 8;
            *reinterpret_cast<ushort8*>(&sm[0][r * LDS_PITCH + kcol]) =
                *reinterpret_cast<const ushort8*>(
                    A + (size_t)(rowblk + r) * K + (size_t)ki * KC + kcol);
            *reinterpret_cast<ushort8*>(&sm[2][r * LDS_PITCH + kcol]) =
                *reinterpret_cast<const ushort8*>(
                    B + (size_t)(colblk + r) * K + (size_t)ki * KC + kcol);
        }
        __syncthreads();
#endif

#pragma unroll
        for (int kk = 0; kk < KC; kk += 32) {
            FragAB a[2], b[4];
            // A 16x32 fragment: lanes 0-15 K [0..7]+[16..23], lanes 16-31
            // K [8..15]+[24..31]  (ISA 7.12.2 A-matrix table)
#pragma unroll
            for (int rt = 0; rt < 2; ++rt) {
                const int r = warow + rt * 16 + l16;
                a[rt].h2[0] = *reinterpret_cast<const ushort8*>(
                    &sm[cur][r * LDS_PITCH + kk + 8 * kh]);
                a[rt].h2[1] = *reinterpret_cast<const ushort8*>(
                    &sm[cur][r * LDS_PITCH + kk + 16 + 8 * kh]);
            }
            // B 32x16 fragment, lane = column: lanes 0-15 K [0..7]+[16..23],
            // lanes 16-31 K [8..15]+[24..31] (same order as A)
#pragma unroll
            for (int nt = 0; nt < 4; ++nt) {
                const int r = wacol + nt * 16 + l16;
                b[nt].h2[0] = *reinterpret_cast<const ushort8*>(
                    &sm[2 + cur][r * LDS_PITCH + kk + 8 * kh]);
                b[nt].h2[1] = *reinterpret_cast<const ushort8*>(
                    &sm[2 + cur][r * LDS_PITCH + kk + 16 + 8 * kh]);
            }
#pragma unroll
            for (int rt = 0; rt < 2; ++rt)
#pragma unroll
                for (int nt = 0; nt < 4; ++nt)
                    acc[rt][nt] = __builtin_amdgcn_wmma_f32_16x16x32_bf16(
                        false, a[rt].bf, false, b[nt].bf,
                        (short)0, acc[rt][nt], false, false);
        }
        __syncthreads();               // everyone done reading before overwrite
    }

    // C/D layout: VGPR v, lanes 0-15 -> row v, lanes 16-31 -> row v+8
#pragma unroll
    for (int rt = 0; rt < 2; ++rt) {
#pragma unroll
        for (int nt = 0; nt < 4; ++nt) {
#pragma unroll
            for (int v = 0; v < 8; ++v) {
                const int r  = rowblk + warow + rt * 16 + v + 8 * kh;
                const int cc = c_col_off + colblk + wacol + nt * 16 + l16;
                if (EPI == 0) {
                    ((float*)C)[(size_t)r * ldc + cc] = acc[rt][nt][v] * alpha;
                } else if (EPI == 1) {
                    ((unsigned short*)C)[(size_t)r * ldc + cc] = f2bf_rn(acc[rt][nt][v]);
                } else {
                    ((float*)C)[(size_t)r * ldc + cc] = tanhf(acc[rt][nt][v] + bias[cc]);
                }
            }
        }
    }
}

// ---------------------------------------------------------------------------
// Double softmax with mask-multiply: one block (256 thr) per row of 4096.
// ---------------------------------------------------------------------------
__global__ void double_softmax_mask(const float* __restrict__ S,
                                    const float* __restrict__ maskF,
                                    const unsigned char* __restrict__ maskB,
                                    unsigned short* __restrict__ Aout) {
    __shared__ float red[256];
    const int row = blockIdx.x;
    const int tid = threadIdx.x;
    const int n   = NSIMP;

    float x[16];
    const float* srow = S + (size_t)row * n;
#pragma unroll
    for (int j = 0; j < 16; ++j) x[j] = srow[tid + j * 256];

    // softmax #1
    float m = -__FLT_MAX__;
#pragma unroll
    for (int j = 0; j < 16; ++j) m = fmaxf(m, x[j]);
    red[tid] = m; __syncthreads();
    for (int off = 128; off > 0; off >>= 1) {
        if (tid < off) red[tid] = fmaxf(red[tid], red[tid + off]);
        __syncthreads();
    }
    m = red[0]; __syncthreads();

    float s = 0.0f;
#pragma unroll
    for (int j = 0; j < 16; ++j) { x[j] = __expf(x[j] - m); s += x[j]; }
    red[tid] = s; __syncthreads();
    for (int off = 128; off > 0; off >>= 1) {
        if (tid < off) red[tid] += red[tid + off];
        __syncthreads();
    }
    float inv = 1.0f / red[0]; __syncthreads();

    // mask multiply (mask is a 0/1 indicator)
    if (maskB) {
        const unsigned char* mrow = maskB + (size_t)row * n;
#pragma unroll
        for (int j = 0; j < 16; ++j)
            x[j] = mrow[tid + j * 256] ? x[j] * inv : 0.0f;
    } else {
        const float* mrow = maskF + (size_t)row * n;
#pragma unroll
        for (int j = 0; j < 16; ++j)
            x[j] = (mrow[tid + j * 256] != 0.0f) ? x[j] * inv : 0.0f;
    }

    // softmax #2
    m = -__FLT_MAX__;
#pragma unroll
    for (int j = 0; j < 16; ++j) m = fmaxf(m, x[j]);
    red[tid] = m; __syncthreads();
    for (int off = 128; off > 0; off >>= 1) {
        if (tid < off) red[tid] = fmaxf(red[tid], red[tid + off]);
        __syncthreads();
    }
    m = red[0]; __syncthreads();

    s = 0.0f;
#pragma unroll
    for (int j = 0; j < 16; ++j) { x[j] = __expf(x[j] - m); s += x[j]; }
    red[tid] = s; __syncthreads();
    for (int off = 128; off > 0; off >>= 1) {
        if (tid < off) red[tid] += red[tid + off];
        __syncthreads();
    }
    inv = 1.0f / red[0];

    unsigned short* arow = Aout + (size_t)row * n;
#pragma unroll
    for (int j = 0; j < 16; ++j) arow[tid + j * 256] = f2bf_rn(x[j] * inv);
}

// ---------------------------------------------------------------------------
// Orchestration
// ---------------------------------------------------------------------------
extern "C" void kernel_launch(void* const* d_in, const int* in_sizes, int n_in,
                              void* d_out, int out_size, void* d_ws, size_t ws_size,
                              hipStream_t stream) {
    const float* L      = (const float*)d_in[0];
    const float* H      = (const float*)d_in[1];
    const float* B_low  = (const float*)d_in[2];
    const float* H_low  = (const float*)d_in[3];
    const float* B_high = (const float*)d_in[4];
    const float* H_high = (const float*)d_in[5];
    const float* W      = (const float*)d_in[6];
    const float* bias   = (const float*)d_in[7];
    float*       out    = (float*)d_out;

    const int N = NSIMP, D = DFEAT;
    const float scale = 0.03125f; // 1/sqrt(1024)

    // Workspace layout (~166 MB total)
    char* ws = (char*)d_ws;
    unsigned short* Hq16  = (unsigned short*)(ws);                        //  8 MB [N,D]
    unsigned short* KV16  = (unsigned short*)(ws + (size_t)8   * 1048576);//  8 MB [N,D]
    unsigned short* KVT16 = (unsigned short*)(ws + (size_t)16  * 1048576);//  8 MB [D,N]
    float*          S     = (float*)         (ws + (size_t)24  * 1048576);// 64 MB [N,N]
    unsigned short* A16   = (unsigned short*)(ws + (size_t)88  * 1048576);// 32 MB [N,N]
    unsigned short* H4b   = (unsigned short*)(ws + (size_t)120 * 1048576);// 24 MB [N,3D]
    unsigned short* W16   = (unsigned short*)(ws + (size_t)144 * 1048576);//  6 MB [D,3D]
    unsigned char*  MTb   = (unsigned char*) (ws + (size_t)150 * 1048576);// 16 MB [N,N]

    // One-time conversions
    cvt_f32_to_bf16<<<2048, 256, 0, stream>>>(H, Hq16, N * D);
    cvt_f32_to_bf16<<<2048, 256, 0, stream>>>(W, W16, D * 3 * D);
    // m_low = (B_low != 0)^T as coalesced byte mask
    mask_transpose<<<(N / 32) * (N / 32), 256, 0, stream>>>(B_low, MTb, N);

    const float* kvsrc[3] = { H, H_low, H_high };
    const float* msrc[3]  = { L, nullptr, B_high };

    const dim3 grid_qk(N / 128, N / 128);   // 32 x 32
    const dim3 grid_av(D / 128, N / 128);   //  8 x 32

    for (int i = 0; i < 3; ++i) {
        const unsigned short* kv16 = (i == 0) ? Hq16 : KV16;
        if (i != 0)
            cvt_f32_to_bf16<<<2048, 256, 0, stream>>>(kvsrc[i], KV16, N * D);
        // KV^T in bf16 so the AV GEMM is NT with contiguous loads
        transpose_cvt_bf16<<<(N / 32) * (D / 32), 256, 0, stream>>>(
            kvsrc[i], KVT16, N, D);

        // S = scale * H * KV^T   (fp32 out)
        gemm_nt_wmma_bf16<0><<<grid_qk, 256, 0, stream>>>(
            Hq16, kv16, S, nullptr, D, N, 0, scale);

        // double softmax with mask -> A16 (bf16)
        double_softmax_mask<<<N, 256, 0, stream>>>(
            S, msrc[i], (i == 1) ? MTb : nullptr, A16);

        // H_i = A * KV = A * (KV^T)^T  (NT with B = KVT, bf16 out into H4)
        gemm_nt_wmma_bf16<1><<<grid_av, 256, 0, stream>>>(
            A16, KVT16, H4b, nullptr, N, 3 * D, i * D, 1.0f);
    }

    // out = tanh(H4 * W^T + b)   (fp32 out)
    gemm_nt_wmma_bf16<2><<<grid_av, 256, 0, stream>>>(
        H4b, W16, out, bias, 3 * D, D, 0, 1.0f);
}
